// TTLSTM_46170898432399
// MI455X (gfx1250) — matrix operand order repaired and checked by reading, compiled, not run on hardware
//
#include <hip/hip_runtime.h>
#include <math.h>

#define B_   32
#define T_   1024
#define I_   512
#define H_   1024
#define G_   4096
#define R_   16
#define NBLK_SCAN 64

// ---------------------------------------------------------------- types ----
typedef __attribute__((ext_vector_type(16))) __bf16 bf16x16;
typedef __attribute__((ext_vector_type(8)))  float  f32x8;

union FragU { uint4 q[2]; bf16x16 v; };

// A fragment (16x32 bf16, M x K), row-major source, per ISA 7.12.2:
// lanes 0-15: M=lane,  K = k0+0..7  and k0+16..23
// lanes16-31: M=lane-16, K = k0+8..15 and k0+24..31
__device__ __forceinline__ bf16x16 load_frag_a(const __bf16* base, int stride,
                                               int m0, int k0, int lane) {
  int row = m0 + (lane & 15);
  int kk  = k0 + ((lane >> 4) << 3);
  const __bf16* p = base + row * stride + kk;
  FragU f;
  f.q[0] = *(const uint4*)(p);
  f.q[1] = *(const uint4*)(p + 16);
  return f.v;
}

// B fragment (32x16 bf16, K x N), loaded from N-major (transposed) weights:
// lanes 0-15: N=lane,  K = k0+0..7 and k0+16..23 ; lanes 16-31: N=lane-16, K = k0+8..15 and k0+24..31
__device__ __forceinline__ bf16x16 load_frag_b(const __bf16* base, int stride,
                                               int n0, int k0, int lane) {
  int col = n0 + (lane & 15);
  int kk  = k0 + ((lane >> 4) << 3);
  const __bf16* p = base + col * stride + kk;
  FragU f;
  f.q[0] = *(const uint4*)(p);
  f.q[1] = *(const uint4*)(p + 16);
  return f.v;
}

__device__ __forceinline__ f32x8 wmma_bf16(bf16x16 a, bf16x16 b, f32x8 c) {
  return __builtin_amdgcn_wmma_f32_16x16x32_bf16(false, a, false, b,
                                                 (short)0, c, false, false);
}

__device__ __forceinline__ float sigmoidf_(float x) {
  return 1.0f / (1.0f + __expf(-x));
}

// ------------------------------------------------------------- kernels ----

// Bias sum, zero both h double-buffers, init grid barrier.
__global__ void k_prep(const float* __restrict__ bih, const float* __restrict__ bhh,
                       float* __restrict__ bsum, __bf16* __restrict__ hbuf,
                       unsigned* __restrict__ bar) {
  int i = blockIdx.x * blockDim.x + threadIdx.x;
  if (i < G_) bsum[i] = bih[i] + bhh[i];
  for (int j = i; j < 2 * B_ * H_; j += gridDim.x * blockDim.x)
    hbuf[j] = (__bf16)0.0f;
  if (i == 0) { bar[0] = 0u; bar[1] = 0u; }
}

// Rank-16 TT reconstruction, stored TRANSPOSED (N-major) in bf16:
// wT[n*M + m] = sum_c w0[m*R + c] * w1[c*N + n]
__global__ void k_recon(const float* __restrict__ w0, const float* __restrict__ w1,
                        __bf16* __restrict__ wT, int M, int N) {
  int idx = blockIdx.x * blockDim.x + threadIdx.x;
  if (idx >= M * N) return;
  int n = idx / M;
  int m = idx - n * M;
  float s = 0.0f;
#pragma unroll
  for (int c = 0; c < R_; ++c) s += w0[m * R_ + c] * w1[c * N + n];
  wT[idx] = (__bf16)s;
}

__global__ void k_cvt(const float* __restrict__ x, __bf16* __restrict__ xb, int n) {
  int i = blockIdx.x * blockDim.x + threadIdx.x;
  if (i < n) xb[i] = (__bf16)x[i];
}

// gx(fp32, 32768 x 4096) = x_bf16(32768 x 512) @ Wih + bias_sum
// One wave computes a 32(M) x 64(N) block: 8 f32 acc tiles.
// Per k-chunk: 2 A frags (4 loads) + 4 B frags (8 loads) feed 8 WMMAs.
__global__ void __launch_bounds__(256) k_gemm_gx(const __bf16* __restrict__ xb,
                                                 const __bf16* __restrict__ wihT,
                                                 const float* __restrict__ bsum,
                                                 float* __restrict__ gx) {
  int lane = threadIdx.x & 31;
  int wid  = blockIdx.x * 8 + (threadIdx.x >> 5);
  int ng = wid & 63;        // 64 N-groups of 64 cols
  int mp = wid >> 6;        // 1024 M-pairs (32 rows each)
  int m0 = mp << 5;
  int n0 = ng << 6;

  f32x8 zero = {0.f, 0.f, 0.f, 0.f, 0.f, 0.f, 0.f, 0.f};
  f32x8 acc[2][4];
#pragma unroll
  for (int mt = 0; mt < 2; ++mt)
#pragma unroll
    for (int g = 0; g < 4; ++g) acc[mt][g] = zero;

  for (int k0 = 0; k0 < I_; k0 += 32) {
    bf16x16 a0 = load_frag_a(xb, I_, m0,      k0, lane);
    bf16x16 a1 = load_frag_a(xb, I_, m0 + 16, k0, lane);
#pragma unroll
    for (int g = 0; g < 4; ++g) {
      bf16x16 b = load_frag_b(wihT, I_, n0 + g * 16, k0, lane);
      acc[0][g] = wmma_bf16(a0, b, acc[0][g]);
      acc[1][g] = wmma_bf16(a1, b, acc[1][g]);
    }
  }
#pragma unroll
  for (int mt = 0; mt < 2; ++mt) {
    int rbase = m0 + mt * 16 + ((lane >> 4) << 3);
#pragma unroll
    for (int g = 0; g < 4; ++g) {
      int col = n0 + g * 16 + (lane & 15);
      float bv = bsum[col];
#pragma unroll
      for (int r = 0; r < 8; ++r)
        gx[(size_t)(rbase + r) * G_ + col] = acc[mt][g][r] + bv;
    }
  }
}

// Device-scope barrier across the 64 persistent scan workgroups.
__device__ __forceinline__ void grid_barrier(volatile unsigned* cnt,
                                             volatile unsigned* gen,
                                             unsigned nblk) {
  __syncthreads();
  __threadfence();
  if (threadIdx.x == 0) {
    unsigned g = *gen;
    unsigned prev = atomicAdd((unsigned*)cnt, 1u);
    if (prev == nblk - 1u) {
      *cnt = 0u;
      __threadfence();
      *gen = g + 1u;
    } else {
      while (*gen == g) { __builtin_amdgcn_s_sleep(2); }
    }
  }
  __threadfence();
  __syncthreads();
}

// Persistent recurrent scan. Block b owns hidden columns [16b, 16b+16).
// Wave w (of 8) computes the K-slice [128w, 128w+128) of h @ Whh for the
// 4 gate column groups (2 M-tiles x 4 gates = 8 WMMA acc tiles per wave).
// The 16 Whh B-fragments per wave are LOOP-INVARIANT and are preloaded into
// registers (128 VGPRs) before the time loop, so the per-step critical path
// is just 8 loads of the L2-resident h buffer + 32 WMMAs + LDS reduce.
__global__ void __launch_bounds__(256, 1) k_scan(const float* __restrict__ gx,
                                                 const __bf16* __restrict__ whhT,
                                                 __bf16* __restrict__ hbuf,
                                                 float* __restrict__ out,
                                                 unsigned* bar) {
  __shared__ float part[4 * 2048];   // 32 KB partial gate tiles
  __shared__ float gfin[2048];       //  8 KB reduced gates (4 x 32 x 16)
  __shared__ float creg[B_ * 16];    //  2 KB persistent cell state

  const int tid  = threadIdx.x;
  const int lane = tid & 31;
  const int w    = tid >> 5;
  const int col0 = blockIdx.x * 16;
  const int kbase = w * 128;
  const size_t hn_off = (size_t)B_ * T_ * H_;
  const size_t cn_off = hn_off + (size_t)B_ * H_;

  for (int i = tid; i < B_ * 16; i += 256) creg[i] = 0.0f;
  __syncthreads();

  // Preload the time-invariant Whh fragments for this wave's K-slice.
  bf16x16 bfrag[4][4];               // [k-chunk][gate]
#pragma unroll
  for (int kc = 0; kc < 4; ++kc)
#pragma unroll
    for (int g = 0; g < 4; ++g)
      bfrag[kc][g] = load_frag_b(whhT, H_, g * H_ + col0, kbase + kc * 32, lane);

  f32x8 zero = {0.f, 0.f, 0.f, 0.f, 0.f, 0.f, 0.f, 0.f};

  for (int t = 0; t < T_; ++t) {
    const __bf16* hcur  = hbuf + (size_t)(t & 1) * (B_ * H_);
    __bf16*       hnext = hbuf + (size_t)((t + 1) & 1) * (B_ * H_);

    f32x8 acc[2][4];
#pragma unroll
    for (int mt = 0; mt < 2; ++mt)
#pragma unroll
      for (int g = 0; g < 4; ++g) acc[mt][g] = zero;

#pragma unroll
    for (int kc = 0; kc < 4; ++kc) {
      int k0 = kbase + kc * 32;
      bf16x16 a0 = load_frag_a(hcur, H_, 0,  k0, lane);
      bf16x16 a1 = load_frag_a(hcur, H_, 16, k0, lane);
#pragma unroll
      for (int g = 0; g < 4; ++g) {
        acc[0][g] = wmma_bf16(a0, bfrag[kc][g], acc[0][g]);
        acc[1][g] = wmma_bf16(a1, bfrag[kc][g], acc[1][g]);
      }
    }

    // Two-phase partial reduction (keeps static LDS under 64 KB).
    if (w < 4) {
#pragma unroll
      for (int mt = 0; mt < 2; ++mt)
#pragma unroll
        for (int g = 0; g < 4; ++g) {
          int tile = mt * 4 + g;
#pragma unroll
          for (int r = 0; r < 8; ++r)
            part[w * 2048 + tile * 256 + r * 32 + lane] = acc[mt][g][r];
        }
    }
    __syncthreads();
    if (w >= 4) {
#pragma unroll
      for (int mt = 0; mt < 2; ++mt)
#pragma unroll
        for (int g = 0; g < 4; ++g) {
          int tile = mt * 4 + g;
#pragma unroll
          for (int r = 0; r < 8; ++r)
            part[(w - 4) * 2048 + tile * 256 + r * 32 + lane] += acc[mt][g][r];
        }
    }
    __syncthreads();
#pragma unroll
    for (int j = 0; j < 8; ++j) {
      int o = tid * 8 + j;
      gfin[o] = part[o] + part[2048 + o] + part[4096 + o] + part[6144 + o];
    }
    __syncthreads();

    // Column-local LSTM cell update: 512 (m, nl) pairs, 2 per thread.
#pragma unroll
    for (int j = 0; j < 2; ++j) {
      int p  = tid * 2 + j;       // p = m*16 + nl
      int m  = p >> 4;
      int nl = p & 15;
      int mt2 = m >> 4;
      int mm  = m & 15;
      int ln  = ((mm >> 3) << 4) | nl;  // C-layout lane
      int r   = mm & 7;
      int obase = (mt2 * 4) * 256 + r * 32 + ln;
      size_t gxrow = ((size_t)m * T_ + t) * G_;
      int colg = col0 + nl;

      float gi = gfin[obase + 0 * 256] + gx[gxrow + 0 * H_ + colg];
      float gf = gfin[obase + 1 * 256] + gx[gxrow + 1 * H_ + colg];
      float gg = gfin[obase + 2 * 256] + gx[gxrow + 2 * H_ + colg];
      float go = gfin[obase + 3 * 256] + gx[gxrow + 3 * H_ + colg];

      float iv = sigmoidf_(gi);
      float fv = sigmoidf_(gf);
      float gv = tanhf(gg);
      float ov = sigmoidf_(go);
      float c  = fv * creg[p] + iv * gv;
      creg[p]  = c;
      float h  = ov * tanhf(c);

      hnext[(size_t)m * H_ + colg] = (__bf16)h;
      out[((size_t)m * T_ + t) * H_ + colg] = h;
      if (t == T_ - 1) {
        out[hn_off + (size_t)m * H_ + colg] = h;
        out[cn_off + (size_t)m * H_ + colg] = c;
      }
    }

    grid_barrier(&bar[0], &bar[1], NBLK_SCAN);
  }
}

// ------------------------------------------------------------- launcher ----
extern "C" void kernel_launch(void* const* d_in, const int* in_sizes, int n_in,
                              void* d_out, int out_size, void* d_ws, size_t ws_size,
                              hipStream_t stream) {
  (void)in_sizes; (void)n_in; (void)out_size; (void)ws_size;
  const float* x    = (const float*)d_in[0];
  const float* wih0 = (const float*)d_in[1];
  const float* wih1 = (const float*)d_in[2];
  const float* whh0 = (const float*)d_in[3];
  const float* whh1 = (const float*)d_in[4];
  const float* bih  = (const float*)d_in[5];
  const float* bhh  = (const float*)d_in[6];
  float* out = (float*)d_out;
  char*  ws  = (char*)d_ws;

  // Workspace layout (bytes)
  __bf16*   wihT = (__bf16*)  (ws + 0);            //  4 MB  (4096 x 512)
  __bf16*   whhT = (__bf16*)  (ws + 4194304);      //  8 MB  (4096 x 1024)
  float*    bsum = (float*)   (ws + 12582912);     // 16 KB
  __bf16*   hbuf = (__bf16*)  (ws + 12599296);     // 128 KB (2 x 32 x 1024)
  unsigned* bar  = (unsigned*)(ws + 12730368);     // barrier state
  __bf16*   xb   = (__bf16*)  (ws + 12730624);     // 32 MB  (32768 x 512)
  float*    gx   = (float*)   (ws + 46285056);     // 512 MB (32768 x 4096)

  k_prep<<<64, 256, 0, stream>>>(bih, bhh, bsum, hbuf, bar);
  k_recon<<<(I_ * G_) / 256, 256, 0, stream>>>(wih0, wih1, wihT, I_, G_);
  k_recon<<<(H_ * G_) / 256, 256, 0, stream>>>(whh0, whh1, whhT, H_, G_);
  k_cvt<<<(B_ * T_ * I_) / 256, 256, 0, stream>>>(x, xb, B_ * T_ * I_);
  k_gemm_gx<<<8192, 256, 0, stream>>>(xb, wihT, bsum, gx);
  k_scan<<<NBLK_SCAN, 256, 0, stream>>>(gx, whhT, hbuf, out, bar);
}
